// TrajGRUBlock_42125039239263
// MI455X (gfx1250) — matrix operand order repaired and checked by reading, not compiled, not run on hardware
//
#include <hip/hip_runtime.h>
#include <math.h>
#include <stdint.h>

// ---------------------------------------------------------------------------
// TrajGRU block for MI455X (gfx1250, wave32).
// Convs = implicit GEMM on v_wmma_f32_16x16x32_f16.
//  * Weights pre-packed (once) into fp16 WMMA A-fragment order
//      [mtile][chunk][lane][16 halfs] -> per-lane contiguous 32B loads.
//  * Activations in halo-padded NHWC fp16 (2-px zero border): every im2col
//    tap is in-bounds -> unconditional global_load_b128 with immediate
//    offsets, no predication, no zero-fill.
//  * Two pixel-tiles per wave: each A fragment feeds two WMMAs.
// B=4, T=10, C=64, H=W=96, L=5.
// ---------------------------------------------------------------------------

typedef __attribute__((ext_vector_type(16))) _Float16 v16h;
typedef __attribute__((ext_vector_type(8)))  float    v8f;

#define BATCH 4
#define TSTEPS 10
#define CH   64
#define Hd   96
#define Wd   96
#define HW   (Hd * Wd)        // 9216
#define NT   (HW / 16)        // 576 pixel tiles (6 per row)
#define NT2  (NT / 2)         // 288 tile-pairs
#define HALO 2
#define Wp   (Wd + 2 * HALO)  // 100
#define Hp   (Hd + 2 * HALO)  // 100
#define PIMG (Hp * Wp)        // padded image pixels

// A (16x32 f16) fragment K-map: lane holds row M=lane%16; element i:
//   K = (lane>=16 ? 16 : 0) + i   (matches contiguous B fragment)
__device__ __forceinline__ int a_k_of(int lane, int i) {
  return ((lane >= 16) ? 16 : 0) + i;
}

// ---------------------------------------------------------------------------
// Accumulate one conv into two accumulators for pixel tiles n0 and n0+16.
//   act  : interior origin (pixel 0,0) of halo-padded (STRIDE-pitch) NHWC fp16
//   packA: [mtile][q][lane][16] fp16, q = (kh*KW+kw)*(CIN/32) + cc
// B-fragment: lane holds col N=lane%16, K=(lane>=16?16:0)+2v+h -> 16
// consecutive channels; second tile is bbase + 16*CIN (same row: n0%32==0).
// ---------------------------------------------------------------------------
template<int KH, int KW, int PAD, int CIN, int STRIDE>
__device__ __forceinline__ void conv_tile2(const _Float16* __restrict__ act,
                                           const _Float16* __restrict__ packA,
                                           int mtile, int n0, v8f& acc0, v8f& acc1) {
  constexpr int CPC = CIN / 32;
  constexpr int NQ  = KH * KW * CPC;
  const int lane = threadIdx.x & 31;
  const int p  = n0 + (lane & 15);
  const int py = p / Wd, px = p % Wd;
  const _Float16* aptr = packA + ((size_t)mtile * NQ * 32 + lane) * 16;
  const int coff = (lane >= 16) ? 16 : 0;
  const _Float16* bbase =
      act + ((size_t)(py - PAD) * STRIDE + (px - PAD)) * CIN + coff;

#pragma unroll
  for (int kh = 0; kh < KH; ++kh)
#pragma unroll
    for (int kw = 0; kw < KW; ++kw)
#pragma unroll
      for (int cc = 0; cc < CPC; ++cc) {
        const int q = (kh * KW + kw) * CPC + cc;
        v16h afr = *(const v16h*)(aptr + (size_t)q * 512);
        const _Float16* bp = bbase + (size_t)(kh * STRIDE + kw) * CIN + cc * 32;
        v16h b0 = *(const v16h*)bp;
        v16h b1 = *(const v16h*)(bp + 16 * CIN);
        acc0 = __builtin_amdgcn_wmma_f32_16x16x32_f16(
            false, afr, false, b0, (short)0, acc0, false, false);
        acc1 = __builtin_amdgcn_wmma_f32_16x16x32_f16(
            false, afr, false, b1, (short)0, acc1, false, false);
      }
}

// D layout: lane l, vgpr v -> M = v + (l>=16 ? 8 : 0), N = l%16.
__device__ __forceinline__ void store_tile_f32(float* __restrict__ out,
                                               const float* __restrict__ bias,
                                               int Cout, int m0, int n0, v8f acc) {
  const int lane = threadIdx.x & 31;
  const int n = n0 + (lane & 15);
#pragma unroll
  for (int v = 0; v < 8; ++v) {
    int m = m0 + v + ((lane >= 16) ? 8 : 0);
    if (m < Cout) out[(size_t)m * HW + n] = acc[v] + bias[m];
  }
}

// ------------------------------ pre-pass kernels ---------------------------

__global__ void k_zero4(float* __restrict__ p, int n) {    // also zeros halves
  int i = blockIdx.x * blockDim.x + threadIdx.x;
  if (i < n) p[i] = 0.0f;
}

// Pack conv weights (Cout, Cin, KH, KW) fp32 -> fp16 A-fragments. One-time.
__global__ void k_packw(const float* __restrict__ w, _Float16* __restrict__ pa,
                        int Cout, int Cin, int KH, int KW, int MT) {
  const int NQ = KH * KW * (Cin / 32);
  const int total = MT * NQ * 32;
  int idx = blockIdx.x * blockDim.x + threadIdx.x;
  if (idx >= total) return;
  const int lane  = idx & 31;
  const int q     = (idx >> 5) % NQ;
  const int mtile = idx / (32 * NQ);
  const int CPC = Cin / 32;
  const int cc  = q % CPC;
  const int kk  = q / CPC;
  const int kw  = kk % KW, kh = kk / KW;
  const int m   = mtile * 16 + (lane & 15);
  _Float16* dst = pa + (size_t)idx * 16;
#pragma unroll
  for (int i = 0; i < 16; ++i) {
    int cin = cc * 32 + a_k_of(lane, i);
    float v = 0.0f;
    if (m < Cout) v = w[(((size_t)m * Cin + cin) * KH + kh) * KW + kw];
    dst[i] = (_Float16)v;
  }
}

// (nimg, C, H, W) fp32 -> halo-padded (nimg, Hp, Wp, C) fp16 interior.
__global__ void k_nhwc_half(const float* __restrict__ src, _Float16* __restrict__ dst,
                            int C, int total) {
  int idx = blockIdx.x * blockDim.x + threadIdx.x;
  if (idx >= total) return;
  int p = idx % HW;
  int c = (idx / HW) % C;
  int n = idx / (C * HW);
  int y = p / Wd, x = p % Wd;
  dst[((size_t)n * PIMG + (size_t)(y + HALO) * Wp + (x + HALO)) * C + c] =
      (_Float16)src[idx];
}

// ------------------------------- step kernels ------------------------------

// i2h at step t: conv3x3(image t*B+b), 64 -> 192, pad 1 (bakes in the
// reference's (B*T)->(T,B) reshape quirk).  Output fp32 CHW per batch.
__global__ void k_i2h(const _Float16* __restrict__ xh,      // (40, Hp, Wp, 64)
                      const _Float16* __restrict__ pa,
                      const float* __restrict__ bias,
                      float* __restrict__ i2h_t, int t) {
  const int MT = 12;
  int wave = blockIdx.x * (blockDim.x >> 5) + (threadIdx.x >> 5);
  if (wave >= BATCH * MT * NT2) return;              // wave-uniform
  int b  = wave / (MT * NT2);
  int r  = wave % (MT * NT2);
  int mt = r / NT2, nt = r % NT2;
  const _Float16* act = xh + ((size_t)(t * BATCH + b) * PIMG + HALO * Wp + HALO) * CH;
  v8f a0 = {}, a1 = {};
  conv_tile2<3, 3, 1, CH, Wp>(act, pa, mt, nt * 32, a0, a1);
  float* ob = i2h_t + (size_t)b * 192 * HW;
  store_tile_f32(ob, bias, 192, mt * 16, nt * 32, a0);
  store_tile_f32(ob, bias, 192, mt * 16, nt * 32 + 16, a1);
}

// f1 = tanh(conv5x5(x_t) + conv5x5(h)), 64 -> 32. Output halo-padded NHWC fp16.
__global__ void k_f1(const _Float16* __restrict__ xh, const _Float16* __restrict__ hh,
                     const _Float16* __restrict__ pa_x, const _Float16* __restrict__ pa_h,
                     const float* __restrict__ b1, const float* __restrict__ b2,
                     _Float16* __restrict__ f1h, int t) {
  const int MT = 2;
  int wave = blockIdx.x * (blockDim.x >> 5) + (threadIdx.x >> 5);
  if (wave >= BATCH * MT * NT2) return;
  int b  = wave / (MT * NT2);
  int r  = wave % (MT * NT2);
  int mt = r / NT2, nt = r % NT2;
  const _Float16* xt = xh + (((size_t)b * TSTEPS + t) * PIMG + HALO * Wp + HALO) * CH;
  const _Float16* hb = hh + ((size_t)b * PIMG + HALO * Wp + HALO) * CH;
  v8f a0 = {}, a1 = {};
  conv_tile2<5, 5, 2, CH, Wp>(xt, pa_x, mt, nt * 32, a0, a1);
  conv_tile2<5, 5, 2, CH, Wp>(hb, pa_h, mt, nt * 32, a0, a1);
  const int lane = threadIdx.x & 31;
  _Float16* ob = f1h + ((size_t)b * PIMG + HALO * Wp + HALO) * 32;
#pragma unroll
  for (int v = 0; v < 8; ++v) {
    int m = mt * 16 + v + ((lane >= 16) ? 8 : 0);
    int p0 = nt * 32 + (lane & 15);
    int p1 = p0 + 16;
    ob[((size_t)(p0 / Wd) * Wp + (p0 % Wd)) * 32 + m] = (_Float16)tanhf(a0[v] + b1[m] + b2[m]);
    ob[((size_t)(p1 / Wd) * Wp + (p1 % Wd)) * 32 + m] = (_Float16)tanhf(a1[v] + b1[m] + b2[m]);
  }
}

// flows = conv5x5(f1), 32 -> 10 (one masked M-tile). Output fp32 CHW.
__global__ void k_flows(const _Float16* __restrict__ f1h,
                        const _Float16* __restrict__ pa,
                        const float* __restrict__ bias, float* __restrict__ fl) {
  int wave = blockIdx.x * (blockDim.x >> 5) + (threadIdx.x >> 5);
  if (wave >= BATCH * NT2) return;
  int b = wave / NT2, nt = wave % NT2;
  const _Float16* act = f1h + ((size_t)b * PIMG + HALO * Wp + HALO) * 32;
  v8f a0 = {}, a1 = {};
  conv_tile2<5, 5, 2, 32, Wp>(act, pa, 0, nt * 32, a0, a1);
  float* ob = fl + (size_t)b * 10 * HW;
  store_tile_f32(ob, bias, 10, 0, nt * 32, a0);
  store_tile_f32(ob, bias, 10, 0, nt * 32 + 16, a1);
}

// Bilinear warp of h (padded NHWC fp16) by -flow_l; output NHWC (HW,320) fp16.
__global__ void k_warp(const _Float16* __restrict__ hh,
                       const float* __restrict__ fl,
                       _Float16* __restrict__ wrp) {
  int idx = blockIdx.x * blockDim.x + threadIdx.x;   // (b, l, pixel)
  if (idx >= BATCH * 5 * HW) return;
  int p  = idx % HW;
  int bl = idx / HW;
  int l  = bl % 5, b = bl / 5;
  int px = p % Wd, py = p / Wd;
  const float* f = fl + ((size_t)b * 10 + (size_t)l * 2) * HW;
  float flx = f[p], fly = f[HW + p];
  // _warp(prev_h, -f): fx = (x - flow_x) * W/(W-1) - 0.5
  float fx = ((float)px - flx) * ((float)Wd / (float)(Wd - 1)) - 0.5f;
  float fy = ((float)py - fly) * ((float)Hd / (float)(Hd - 1)) - 0.5f;
  float x0f = floorf(fx), y0f = floorf(fy);
  float wx = fx - x0f, wy = fy - y0f;
  int x0 = (int)x0f, y0 = (int)y0f, x1 = x0 + 1, y1 = y0 + 1;
  float m00 = (x0 >= 0 && x0 < Wd && y0 >= 0 && y0 < Hd) ? 1.0f : 0.0f;
  float m01 = (x1 >= 0 && x1 < Wd && y0 >= 0 && y0 < Hd) ? 1.0f : 0.0f;
  float m10 = (x0 >= 0 && x0 < Wd && y1 >= 0 && y1 < Hd) ? 1.0f : 0.0f;
  float m11 = (x1 >= 0 && x1 < Wd && y1 >= 0 && y1 < Hd) ? 1.0f : 0.0f;
  int cx0 = min(max(x0, 0), Wd - 1), cx1 = min(max(x1, 0), Wd - 1);
  int cy0 = min(max(y0, 0), Hd - 1), cy1 = min(max(y1, 0), Hd - 1);
  float w00 = (1.0f - wx) * (1.0f - wy) * m00;
  float w01 = wx * (1.0f - wy) * m01;
  float w10 = (1.0f - wx) * wy * m10;
  float w11 = wx * wy * m11;
  const _Float16* hb = hh + (size_t)b * PIMG * CH;
  const _Float16* h00 = hb + ((size_t)(cy0 + HALO) * Wp + cx0 + HALO) * CH;
  const _Float16* h01 = hb + ((size_t)(cy0 + HALO) * Wp + cx1 + HALO) * CH;
  const _Float16* h10 = hb + ((size_t)(cy1 + HALO) * Wp + cx0 + HALO) * CH;
  const _Float16* h11 = hb + ((size_t)(cy1 + HALO) * Wp + cx1 + HALO) * CH;
  _Float16* ob = wrp + ((size_t)b * HW + p) * 320 + l * CH;
#pragma unroll 8
  for (int c = 0; c < CH; ++c) {
    float v = (float)h00[c] * w00 + (float)h01[c] * w01 +
              (float)h10[c] * w10 + (float)h11[c] * w11;
    ob[c] = (_Float16)v;
  }
}

// h2h = conv1x1(warped NHWC fp16, unpadded), 320 -> 192. Output fp32 CHW.
__global__ void k_h2h(const _Float16* __restrict__ wrp,
                      const _Float16* __restrict__ pa,
                      const float* __restrict__ bias, float* __restrict__ h2h) {
  const int MT = 12;
  int wave = blockIdx.x * (blockDim.x >> 5) + (threadIdx.x >> 5);
  if (wave >= BATCH * MT * NT2) return;
  int b  = wave / (MT * NT2);
  int r  = wave % (MT * NT2);
  int mt = r / NT2, nt = r % NT2;
  v8f a0 = {}, a1 = {};
  conv_tile2<1, 1, 0, 320, Wd>(wrp + (size_t)b * HW * 320, pa, mt, nt * 32, a0, a1);
  float* ob = h2h + (size_t)b * 192 * HW;
  store_tile_f32(ob, bias, 192, mt * 16, nt * 32, a0);
  store_tile_f32(ob, bias, 192, mt * 16, nt * 32 + 16, a1);
}

// GRU gates -> nh; update h (fp32 CHW in-place + padded fp16 NHWC), outs[b][t].
__global__ void k_gates(const float* __restrict__ i2h_t,
                        const float* __restrict__ h2h,
                        float* __restrict__ hf,            // fp32 (B,64,H,W)
                        _Float16* __restrict__ hh,         // fp16 (B,Hp,Wp,64)
                        float* __restrict__ out, int t) {
  int idx = blockIdx.x * blockDim.x + threadIdx.x;
  if (idx >= BATCH * CH * HW) return;
  int p = idx % HW;
  int c = (idx / HW) % CH;
  int b = idx / (CH * HW);
  size_t g = (size_t)b * 192 * HW + (size_t)c * HW + p;
  float r = 1.0f / (1.0f + expf(-(i2h_t[g] + h2h[g])));
  float u = 1.0f / (1.0f + expf(-(i2h_t[g + (size_t)CH * HW] + h2h[g + (size_t)CH * HW])));
  float m = tanhf(i2h_t[g + (size_t)2 * CH * HW] + r * h2h[g + (size_t)2 * CH * HW]);
  size_t hi = (size_t)b * CH * HW + (size_t)c * HW + p;
  float nh = u * hf[hi] + (1.0f - u) * m;
  hf[hi] = nh;
  int y = p / Wd, x = p % Wd;
  hh[((size_t)b * PIMG + (size_t)(y + HALO) * Wp + (x + HALO)) * CH + c] = (_Float16)nh;
  out[(((size_t)b * TSTEPS + t) * CH + c) * HW + p] = nh;   // outs (B,T,C,H,W)
}

// ------------------------------- launcher ----------------------------------

extern "C" void kernel_launch(void* const* d_in, const int* in_sizes, int n_in,
                              void* d_out, int out_size, void* d_ws, size_t ws_size,
                              hipStream_t stream) {
  const float* inputs  = (const float*)d_in[0];
  const float* i2h_w   = (const float*)d_in[1];
  const float* i2h_b   = (const float*)d_in[2];
  const float* i2f_w   = (const float*)d_in[3];
  const float* i2f_b   = (const float*)d_in[4];
  const float* h2f_w   = (const float*)d_in[5];
  const float* h2f_b   = (const float*)d_in[6];
  const float* flows_w = (const float*)d_in[7];
  const float* flows_b = (const float*)d_in[8];
  const float* ret_w   = (const float*)d_in[9];
  const float* ret_b   = (const float*)d_in[10];
  float* out = (float*)d_out;

  // 256B-aligned workspace carve-up.
  uintptr_t base = (uintptr_t)d_ws;
  auto alloc = [&](size_t bytes) -> void* {
    void* p = (void*)base;
    base += (bytes + 255) & ~(size_t)255;
    return p;
  };
  const size_t hN   = (size_t)BATCH * CH * HW;
  const size_t xhN  = (size_t)BATCH * TSTEPS * PIMG * CH;   // halfs
  const size_t hhN  = (size_t)BATCH * PIMG * CH;            // halfs
  const size_t f1N  = (size_t)BATCH * PIMG * 32;            // halfs
  float*     hf    = (float*)    alloc(hN * 4);             // h fp32 CHW
  _Float16*  hh    = (_Float16*) alloc(hhN * 2);            // h fp16 padded NHWC
  _Float16*  xh    = (_Float16*) alloc(xhN * 2);            // inputs padded NHWC
  _Float16*  f1h   = (_Float16*) alloc(f1N * 2);
  float*     i2h_t = (float*)    alloc((size_t)BATCH * 192 * HW * 4);
  float*     fl    = (float*)    alloc((size_t)BATCH * 10 * HW * 4);
  _Float16*  wrp   = (_Float16*) alloc((size_t)BATCH * 320 * HW * 2);
  float*     h2h   = (float*)    alloc((size_t)BATCH * 192 * HW * 4);
  _Float16*  pa_i2h = (_Float16*)alloc((size_t)12 * 18 * 512 * 2);
  _Float16*  pa_i2f = (_Float16*)alloc((size_t)2  * 50 * 512 * 2);
  _Float16*  pa_h2f = (_Float16*)alloc((size_t)2  * 50 * 512 * 2);
  _Float16*  pa_fl  = (_Float16*)alloc((size_t)1  * 25 * 512 * 2);
  _Float16*  pa_ret = (_Float16*)alloc((size_t)12 * 10 * 512 * 2);
  if (base - (uintptr_t)d_ws > ws_size) return;

  const int TPB = 128;                 // 4 wave32s per block
  const int WPB = TPB / 32;
  auto wgrid = [&](int waves) { return (waves + WPB - 1) / WPB; };
  auto tgrid = [&](long n)    { return (int)((n + TPB - 1) / TPB); };

  // ---- one-time pre-passes: zero padded buffers (borders stay 0), convert,
  //      pack weights ----
  k_zero4<<<tgrid(hN), TPB, 0, stream>>>(hf, (int)hN);
  k_zero4<<<tgrid(hhN / 2), TPB, 0, stream>>>((float*)hh, (int)(hhN / 2));
  k_zero4<<<tgrid(xhN / 2), TPB, 0, stream>>>((float*)xh, (int)(xhN / 2));
  k_zero4<<<tgrid(f1N / 2), TPB, 0, stream>>>((float*)f1h, (int)(f1N / 2));
  k_nhwc_half<<<tgrid((long)BATCH * TSTEPS * CH * HW), TPB, 0, stream>>>(
      inputs, xh, CH, BATCH * TSTEPS * CH * HW);
  k_packw<<<tgrid(12 * 18 * 32), TPB, 0, stream>>>(i2h_w,   pa_i2h, 192, 64, 3, 3, 12);
  k_packw<<<tgrid(2  * 50 * 32), TPB, 0, stream>>>(i2f_w,   pa_i2f, 32, 64, 5, 5, 2);
  k_packw<<<tgrid(2  * 50 * 32), TPB, 0, stream>>>(h2f_w,   pa_h2f, 32, 64, 5, 5, 2);
  k_packw<<<tgrid(1  * 25 * 32), TPB, 0, stream>>>(flows_w, pa_fl,  10, 32, 5, 5, 1);
  k_packw<<<tgrid(12 * 10 * 32), TPB, 0, stream>>>(ret_w,   pa_ret, 192, 320, 1, 1, 12);

  // ---- recurrence ----
  for (int t = 0; t < TSTEPS; ++t) {
    k_i2h  <<<wgrid(BATCH * 12 * NT2), TPB, 0, stream>>>(xh, pa_i2h, i2h_b, i2h_t, t);
    k_f1   <<<wgrid(BATCH * 2  * NT2), TPB, 0, stream>>>(xh, hh, pa_i2f, pa_h2f,
                                                         i2f_b, h2f_b, f1h, t);
    k_flows<<<wgrid(BATCH * NT2), TPB, 0, stream>>>(f1h, pa_fl, flows_b, fl);
    k_warp <<<tgrid((long)BATCH * 5 * HW), TPB, 0, stream>>>(hh, fl, wrp);
    k_h2h  <<<wgrid(BATCH * 12 * NT2), TPB, 0, stream>>>(wrp, pa_ret, ret_b, h2h);
    k_gates<<<tgrid(hN), TPB, 0, stream>>>(i2h_t, h2h, hf, hh, out, t);
  }
  // last_h follows the (B,T,C,H,W) outs block.
  hipMemcpyAsync(out + (size_t)BATCH * TSTEPS * CH * HW, hf,
                 hN * sizeof(float), hipMemcpyDeviceToDevice, stream);
}
